// _ElmoModel_40862318854408
// MI455X (gfx1250) — matrix operand order repaired and checked by reading, not compiled, not run on hardware
//
#include <hip/hip_runtime.h>
#include <hip/hip_bf16.h>

// ELMo forward on gfx1250: all GEMMs via v_wmma_f32_16x16x32_f16.
// B=8, T=32, C=50, N_FILT=2048, PROJ=512, CELL=4096, GATES=16384, out=(3,8,32,1024) f32.

#define B_     8
#define T_     32
#define CC_    50
#define NF_    2048
#define PROJ_  512
#define CELL_  4096
#define GATES_ 16384
#define CLIP_  3.0f

typedef __attribute__((ext_vector_type(16))) _Float16 v16h;
typedef __attribute__((ext_vector_type(8)))  float    v8f;

union V16H { v16h v; _Float16 h[16]; };
union V8F  { v8f  v; float     f[8];  };

__device__ __forceinline__ float sigmoidf_(float x) { return 1.0f / (1.0f + __expf(-x)); }

// ---------------------------------------------------------------------------
// WMMA GEMM: out[M,N] = act( A[M,K] @ W[N,K]^T + bias[N] + Cadd[M,N] )
// Block = 128 threads (4 waves). Block tile = (16*MT) rows x 64 cols.
// Each wave owns one 16-col strip and MT row-tiles; the B fragment is loaded
// once per K-substep and reused for all MT row-tiles (halves W traffic for
// MT=2, doubles WMMA per byte). K staged in 64-wide LDS chunks (K % 64 == 0,
// N % 64 == 0). act: 0 = none, 1 = clip to [-3,3].
// ---------------------------------------------------------------------------
template<int MT>
__global__ void gemm_wmma_kernel(const float* __restrict__ A, int lda,
                                 const float* __restrict__ W, int ldw,
                                 const float* __restrict__ Cadd, long ldc,
                                 const float* __restrict__ bias,
                                 float* __restrict__ out, int ldo,
                                 int M, int N, int K, int act)
{
    const int lane  = threadIdx.x & 31;
    const int wave  = threadIdx.x >> 5;
    const int mTile = blockIdx.y * (16 * MT);
    const int nTile = blockIdx.x * 64 + wave * 16;

    __shared__ _Float16 As[MT][16][64];      // A tiles, [mt][m][k], f16

    V8F acc[MT];
#pragma unroll
    for (int i = 0; i < MT; ++i) acc[i].v = (v8f){};

    const int mA  = lane & 15;               // A/C fragment row group
    const int kHi = lane >> 4;               // 0 or 1 (K-half select)
    const int nB  = nTile + (lane & 15);     // B/C fragment column

    for (int k0 = 0; k0 < K; k0 += 64) {
        __syncthreads();                     // protect As from previous readers
        // stage MT x 16 x 64 A elements, fp32 -> f16 (coalesced: tid -> k)
#pragma unroll
        for (int r = 0; r < MT * 8; ++r) {
            int idx = threadIdx.x + (r << 7);        // 0 .. MT*1024-1
            int mt  = idx >> 10;
            int mm  = (idx >> 6) & 15;
            int kk  = idx & 63;
            int grow = mTile + mt * 16 + mm;
            float v = (grow < M) ? A[(long)grow * lda + k0 + kk] : 0.0f;
            As[mt][mm][kk] = (_Float16)v;
        }
        __syncthreads();

#pragma unroll
        for (int kc = 0; kc < 2; ++kc) {
            // B fragment (this code's contiguous k order: lane -> N, k = e + 16*kHi)
            V16H b;
            const float* wp = W + (long)nB * ldw + k0 + (kc << 5) + (kHi << 4);
#pragma unroll
            for (int e = 0; e < 16; ++e) b.h[e] = (_Float16)wp[e];

#pragma unroll
            for (int mt = 0; mt < MT; ++mt) {
                // A fragment (contiguous k order, same as B: k = e + 16*kHi)
                V16H a;
#pragma unroll
                for (int e = 0; e < 16; ++e) {
                    int kk = (kc << 5) + e + (kHi << 4);
                    a.h[e] = As[mt][mA][kk];
                }
                acc[mt].v = __builtin_amdgcn_wmma_f32_16x16x32_f16(
                                false, a.v, false, b.v, (short)0, acc[mt].v,
                                false, false);
            }
        }
    }

    // Epilogue: C/D layout: row m = mTile + mt*16 + r + 8*kHi, col n = nB.
    float bv = bias ? bias[nB] : 0.0f;
#pragma unroll
    for (int mt = 0; mt < MT; ++mt) {
#pragma unroll
        for (int r = 0; r < 8; ++r) {
            int m = mTile + mt * 16 + r + (kHi << 3);
            if (m < M) {
                float v = acc[mt].f[r] + bv;
                if (Cadd) v += Cadd[(long)m * ldc + nB];
                if (act == 1) v = fminf(CLIP_, fmaxf(-CLIP_, v));
                out[(long)m * ldo + nB] = v;
            }
        }
    }
}

// ---------------------------------------------------------------------------
// Char CNN: one block per token (256 blocks, 256 threads).
// hfeat[tok][f] = relu( max_p conv(emb, Wf)[p] + bias[f] )
// ---------------------------------------------------------------------------
__global__ void char_cnn_kernel(const int* __restrict__ chars,
                                const float* __restrict__ emb,
                                const float* __restrict__ w1, const float* __restrict__ cb1,
                                const float* __restrict__ w2, const float* __restrict__ cb2,
                                const float* __restrict__ w3, const float* __restrict__ cb3,
                                const float* __restrict__ w4, const float* __restrict__ cb4,
                                const float* __restrict__ w5, const float* __restrict__ cb5,
                                const float* __restrict__ w6, const float* __restrict__ cb6,
                                const float* __restrict__ w7, const float* __restrict__ cb7,
                                float* __restrict__ hfeat)
{
    int tok = blockIdx.x;                       // 0..255
    __shared__ float E[CC_][16];                // [pos][chan]
    for (int i = threadIdx.x; i < CC_ * 16; i += blockDim.x) {
        int p = i >> 4, c = i & 15;
        int ch = chars[tok * CC_ + p];
        E[p][c] = emb[ch * 16 + c];
    }
    __syncthreads();

#pragma unroll
    for (int j = 0; j < 8; ++j) {
        int f = threadIdx.x + (j << 8);         // 0..2047
        int width, local; const float* Wt; const float* Bt;
        if      (f <   32) { width = 1; local = f;        Wt = w1; Bt = cb1; }
        else if (f <   64) { width = 2; local = f -   32; Wt = w2; Bt = cb2; }
        else if (f <  128) { width = 3; local = f -   64; Wt = w3; Bt = cb3; }
        else if (f <  256) { width = 4; local = f -  128; Wt = w4; Bt = cb4; }
        else if (f <  512) { width = 5; local = f -  256; Wt = w5; Bt = cb5; }
        else if (f < 1024) { width = 6; local = f -  512; Wt = w6; Bt = cb6; }
        else               { width = 7; local = f - 1024; Wt = w7; Bt = cb7; }
        const float* Wf = Wt + (long)local * 16 * width;  // (16, width) row-major
        float best = -1e30f;
        for (int p = 0; p <= CC_ - width; ++p) {
            float s = 0.0f;
            for (int c = 0; c < 16; ++c)
                for (int k = 0; k < width; ++k)
                    s += E[p + k][c] * Wf[c * width + k];
            best = fmaxf(best, s);
        }
        float y = best + Bt[local];
        hfeat[(long)tok * NF_ + f] = fmaxf(y, 0.0f);
    }
}

// highway gating: h = g*h + (1-g)*relu(lin);  p is (256,4096): [:,0:2048]=lin, [:,2048:]=gate
__global__ void highway_kernel(const float* __restrict__ p, float* __restrict__ h)
{
    int i = blockIdx.x * blockDim.x + threadIdx.x;      // 256*2048
    if (i >= 256 * NF_) return;
    int row = i >> 11, col = i & 2047;
    float lin  = p[(long)row * 4096 + col];
    float gate = p[(long)row * 4096 + 2048 + col];
    float g = sigmoidf_(gate);
    h[i] = g * h[i] + (1.0f - g) * fmaxf(lin, 0.0f);
}

// LSTM cell nonlinearity: g(8,16384) -> c update (masked), u = o*tanh(c_new)
__global__ void lstm_cell_kernel(const float* __restrict__ g,
                                 float* __restrict__ c, float* __restrict__ u,
                                 const int* __restrict__ lengths, int t)
{
    int i = blockIdx.x * blockDim.x + threadIdx.x;      // 8*4096
    if (i >= B_ * CELL_) return;
    int b = i >> 12, j = i & 4095;
    const float* gb = g + (long)b * GATES_;
    float ig = sigmoidf_(gb[j]);
    float fg = sigmoidf_(gb[CELL_ + j]);
    float z  = tanhf(gb[2 * CELL_ + j]);
    float og = sigmoidf_(gb[3 * CELL_ + j]);
    float cold = c[i];
    float cn = fminf(CLIP_, fmaxf(-CLIP_, ig * z + fg * cold));
    u[i] = og * tanhf(cn);
    c[i] = (t < lengths[b]) ? cn : cold;
}

// masked h update + write hs[:, t, :]
__global__ void lstm_update_kernel(const float* __restrict__ hn,
                                   float* __restrict__ h, float* __restrict__ outdir,
                                   const int* __restrict__ lengths, int t)
{
    int i = blockIdx.x * blockDim.x + threadIdx.x;      // 8*512
    if (i >= B_ * PROJ_) return;
    int b = i >> 9, j = i & 511;
    bool mt = (t < lengths[b]);
    float v = hn[i];
    if (mt) h[i] = v;
    outdir[((long)(b * T_ + t)) * PROJ_ + j] = mt ? v : 0.0f;
}

__global__ void fill_kernel(float* __restrict__ p, float v, int n)
{
    int i = blockIdx.x * blockDim.x + threadIdx.x;
    if (i < n) p[i] = v;
}

__global__ void add_kernel(float* __restrict__ y, const float* __restrict__ x, int n)
{
    int i = blockIdx.x * blockDim.x + threadIdx.x;
    if (i < n) y[i] += x[i];
}

// dst(8,32,1024) = concat(f(8,32,512), bw(8,32,512)) along last dim
__global__ void pack_kernel(const float* __restrict__ f, const float* __restrict__ bw,
                            float* __restrict__ dst)
{
    int i = blockIdx.x * blockDim.x + threadIdx.x;      // 8*32*1024
    if (i >= B_ * T_ * 1024) return;
    int row = i >> 10, j = i & 1023;
    dst[i] = (j < 512) ? f[(long)row * 512 + j] : bw[(long)row * 512 + (j - 512)];
}

// ---------------------------------------------------------------------------

static inline void gemm(hipStream_t s, const float* A, int lda,
                        const float* W, int ldw,
                        const float* Cadd, long ldc, const float* bias,
                        float* out, int ldo, int M, int N, int K, int act)
{
    if (M > 16) {
        dim3 grid(N / 64, (M + 31) / 32);
        gemm_wmma_kernel<2><<<grid, 128, 0, s>>>(A, lda, W, ldw, Cadd, ldc, bias,
                                                 out, ldo, M, N, K, act);
    } else {
        dim3 grid(N / 64, 1);
        gemm_wmma_kernel<1><<<grid, 128, 0, s>>>(A, lda, W, ldw, Cadd, ldc, bias,
                                                 out, ldo, M, N, K, act);
    }
}

extern "C" void kernel_launch(void* const* d_in, const int* in_sizes, int n_in,
                              void* d_out, int out_size, void* d_ws, size_t ws_size,
                              hipStream_t stream)
{
    (void)in_sizes; (void)n_in; (void)out_size; (void)ws_size;

    const int*   chars    = (const int*)  d_in[0];
    const int*   lengths  = (const int*)  d_in[1];
    const float* char_emb = (const float*)d_in[2];
    const float* cw[7], *cb[7];
    for (int i = 0; i < 7; ++i) { cw[i] = (const float*)d_in[3 + 2*i]; cb[i] = (const float*)d_in[4 + 2*i]; }
    const float* hw_w   = (const float*)d_in[17];   // (2,4096,2048)
    const float* hw_b   = (const float*)d_in[18];   // (2,4096)
    const float* proj_w = (const float*)d_in[19];   // (512,2048)
    const float* proj_b = (const float*)d_in[20];   // (512)
    const float* Wi     = (const float*)d_in[21];   // (2,2,16384,512)
    const float* Ws     = (const float*)d_in[22];   // (2,2,16384,512)
    const float* lb     = (const float*)d_in[23];   // (2,2,16384)
    const float* Wp     = (const float*)d_in[24];   // (2,2,512,4096)
    float* out = (float*)d_out;                     // (3,8,32,1024)

    // workspace layout (floats)
    float* w = (float*)d_ws;
    float* hfeat = w; w += 256 * NF_;        // char-cnn features
    float* pbuf  = w; w += 256 * 4096;       // highway pre-activation
    float* tok   = w; w += 256 * PROJ_;      // projected tokens (layer-0 input)
    float* xi    = w; w += 256 * GATES_;     // precomputed input gates
    float* hbuf  = w; w += B_ * PROJ_;       // LSTM h state
    float* cbuf  = w; w += B_ * CELL_;       // LSTM c state
    float* gbuf  = w; w += B_ * GATES_;      // gate pre-activations
    float* ubuf  = w; w += B_ * CELL_;       // o * tanh(c)
    float* hnb   = w; w += B_ * PROJ_;       // candidate h
    float* of0   = w; w += 256 * PROJ_;      // layer-0 fwd hidden seq
    float* ob0   = w; w += 256 * PROJ_;      // layer-0 bwd hidden seq
    float* of1   = w; w += 256 * PROJ_;      // layer-1 fwd hidden seq
    float* ob1   = w; w += 256 * PROJ_;      // layer-1 bwd hidden seq

    // 1) char CNN -> hfeat (256,2048)
    char_cnn_kernel<<<256, 256, 0, stream>>>(chars, char_emb,
        cw[0], cb[0], cw[1], cb[1], cw[2], cb[2], cw[3], cb[3],
        cw[4], cb[4], cw[5], cb[5], cw[6], cb[6], hfeat);

    // 2) two highway layers (WMMA GEMM + gating)
    for (int i = 0; i < 2; ++i) {
        gemm(stream, hfeat, NF_, hw_w + (long)i * 4096 * NF_, NF_,
             nullptr, 0, hw_b + (long)i * 4096, pbuf, 4096, 256, 4096, NF_, 0);
        highway_kernel<<<(256 * NF_ + 255) / 256, 256, 0, stream>>>(pbuf, hfeat);
    }

    // 3) projection -> tok (256,512)
    gemm(stream, hfeat, NF_, proj_w, NF_, nullptr, 0, proj_b, tok, PROJ_, 256, PROJ_, NF_, 0);

    // 4) BiLSTM, 2 layers x 2 directions
    float* outs[2][2] = { { of0, ob0 }, { of1, ob1 } };
    const float* xin[2] = { tok, tok };      // current layer inputs (fwd, bwd)

    for (int l = 0; l < 2; ++l) {
        for (int d = 0; d < 2; ++d) {
            long ld_off = (long)(l * 2 + d);
            const float* x = xin[d];
            // xi = x @ Wi^T  (256,16384)
            gemm(stream, x, PROJ_, Wi + ld_off * GATES_ * PROJ_, PROJ_,
                 nullptr, 0, nullptr, xi, GATES_, 256, GATES_, PROJ_, 0);
            // h, c = 0
            fill_kernel<<<(B_ * PROJ_ + 255) / 256, 256, 0, stream>>>(hbuf, 0.0f, B_ * PROJ_);
            fill_kernel<<<(B_ * CELL_ + 255) / 256, 256, 0, stream>>>(cbuf, 0.0f, B_ * CELL_);
            for (int s = 0; s < T_; ++s) {
                int t = (d == 0) ? s : (T_ - 1 - s);
                // g = h @ Ws^T + b + xi[:, t, :]
                gemm(stream, hbuf, PROJ_, Ws + ld_off * GATES_ * PROJ_, PROJ_,
                     xi + (long)t * GATES_, (long)T_ * GATES_,
                     lb + ld_off * GATES_, gbuf, GATES_, B_, GATES_, PROJ_, 0);
                lstm_cell_kernel<<<(B_ * CELL_ + 255) / 256, 256, 0, stream>>>(
                    gbuf, cbuf, ubuf, lengths, t);
                // hn = clip( u @ Wp^T )
                gemm(stream, ubuf, CELL_, Wp + ld_off * PROJ_ * CELL_, CELL_,
                     nullptr, 0, nullptr, hnb, PROJ_, B_, PROJ_, CELL_, 1);
                lstm_update_kernel<<<(B_ * PROJ_ + 255) / 256, 256, 0, stream>>>(
                    hnb, hbuf, outs[l][d], lengths, t);
            }
        }
        if (l > 0) {  // residual for l>0: out += input
            add_kernel<<<(256 * PROJ_ + 255) / 256, 256, 0, stream>>>(outs[l][0], xin[0], 256 * PROJ_);
            add_kernel<<<(256 * PROJ_ + 255) / 256, 256, 0, stream>>>(outs[l][1], xin[1], 256 * PROJ_);
        }
        xin[0] = outs[l][0];
        xin[1] = outs[l][1];
    }

    // 5) assemble output: layer 0 = [tok, tok], layer l+1 = [f_l, bw_l]
    const int LSZ = B_ * T_ * 1024;
    pack_kernel<<<(LSZ + 255) / 256, 256, 0, stream>>>(tok, tok, out);
    pack_kernel<<<(LSZ + 255) / 256, 256, 0, stream>>>(of0, ob0, out + LSZ);
    pack_kernel<<<(LSZ + 255) / 256, 256, 0, stream>>>(of1, ob1, out + 2 * LSZ);
}
